// ESP32Attention_8675833938754
// MI455X (gfx1250) — compile-verified, matrix operand order checked by reading, not run on hardware
//
#include <hip/hip_runtime.h>
#include <hip/hip_bf16.h>

// ---------------------------------------------------------------------------
// MHA block for [4, 2048, 1024], 16 heads, head_dim 64, on gfx1250 (CDNA5).
// bf16 WMMA (v_wmma_f32_16x16x32_bf16) for all three GEMM stages +
// flash attention with online softmax. Compute-bound (~137 GFLOP vs ~80MB).
// GEMM: 128x128 tile, 8 waves, K-step 32 (8 WMMA / wave / K-step).
// Attention: 128-q-row tile, 8 waves, 64-key blocks.
// ---------------------------------------------------------------------------

typedef __attribute__((ext_vector_type(16))) __bf16 v16bf;
typedef __attribute__((ext_vector_type(8)))  float  v8f;

union FragU {
    v16bf    v;
    unsigned u[8];
};

__device__ __forceinline__ unsigned short f32_to_bf16(float f) {
    unsigned u = __float_as_uint(f);
    unsigned r = (u + 0x7FFFu + ((u >> 16) & 1u)) >> 16;   // RNE
    return (unsigned short)r;
}

// K-pair base inside a bf16 A/B fragment (ISA 7.12.2, 16-bit 16x32 layout):
// lanes 0-15 hold K = {0..7, 16..23}, lanes 16-31 hold K = {8..15, 24..31},
// two consecutive K values per VGPR -> one aligned u32 read per pair, and the
// 8 pairs per half form two contiguous 16B runs (compiler -> ds_load_b128).
__device__ __forceinline__ int frag_kk(int p, int half) {
    return (p < 4) ? (half * 8 + 2 * p) : (16 + half * 8 + 2 * (p - 4));
}

// ---------------------------------------------------------------------------
// fp32 -> bf16 conversion
// ---------------------------------------------------------------------------
__global__ void cvt_f32_bf16_kernel(const float* __restrict__ src,
                                    unsigned short* __restrict__ dst, int n) {
    int i = blockIdx.x * blockDim.x + threadIdx.x;
    if (i < n) dst[i] = f32_to_bf16(src[i]);
}

// ---------------------------------------------------------------------------
// GEMM 1: QKV projection. C[8192,3072] = Xbf[8192,1024] @ Wbf^T + bias,
// scattered into Q/K/V buffers [b(4)][h(16)][s(2048)][hd(64)] bf16.
// Block: 256 thr (8 waves). Tile 128(M) x 128(N), K-step 32.
// Wave (wr = wave&3, wc = wave>>2) owns rows wr*32..+31, cols wc*64..+63.
// ---------------------------------------------------------------------------
__global__ __launch_bounds__(256) void gemm_qkv_kernel(
    const unsigned short* __restrict__ A,     // [8192][1024] bf16
    const unsigned short* __restrict__ W,     // [3072][1024] bf16
    const float* __restrict__ bias,           // [3072]
    unsigned short* __restrict__ Qb,
    unsigned short* __restrict__ Kb,
    unsigned short* __restrict__ Vb)
{
    __shared__ unsigned short As[128][32];
    __shared__ unsigned short Bs[128][32];

    const int tid  = threadIdx.x;
    const int wave = tid >> 5;
    const int lane = tid & 31;
    const int half = lane >> 4;
    const int lidx = lane & 15;
    const int wr   = wave & 3;
    const int wc   = wave >> 2;
    const int mBase = blockIdx.x * 128;
    const int nBase = blockIdx.y * 128;

    v8f acc[2][4];
    for (int i = 0; i < 2; ++i)
        for (int j = 0; j < 4; ++j)
            for (int e = 0; e < 8; ++e) acc[i][j][e] = 0.0f;

    const uint4* Ag4 = (const uint4*)A;   // global row = 128 uint4
    const uint4* Wg4 = (const uint4*)W;

    for (int kb = 0; kb < 1024; kb += 32) {
        __syncthreads();
        // Stage 128x32 bf16 A and B tiles (512 uint4 each, 2 per thread).
#pragma unroll
        for (int i = 0; i < 2; ++i) {
            int s = i * 256 + tid;
            int r = s >> 2, c = s & 3;
            ((uint4*)As)[s] = Ag4[(size_t)(mBase + r) * 128 + (kb >> 3) + c];
            ((uint4*)Bs)[s] = Wg4[(size_t)(nBase + r) * 128 + (kb >> 3) + c];
        }
        __syncthreads();

        FragU af[2];
#pragma unroll
        for (int i = 0; i < 2; ++i) {
            const int arow = wr * 32 + i * 16 + lidx;
#pragma unroll
            for (int p = 0; p < 8; ++p)
                af[i].u[p] = *(const unsigned*)&As[arow][frag_kk(p, half)];
        }
#pragma unroll
        for (int j = 0; j < 4; ++j) {
            FragU bf;
            const int brow = wc * 64 + j * 16 + lidx;
#pragma unroll
            for (int p = 0; p < 8; ++p)
                bf.u[p] = *(const unsigned*)&Bs[brow][frag_kk(p, half)];
#pragma unroll
            for (int i = 0; i < 2; ++i)
                acc[i][j] = __builtin_amdgcn_wmma_f32_16x16x32_bf16(
                    false, af[i].v, false, bf.v, (short)0, acc[i][j], false, false);
        }
    }

    // Epilogue: +bias, scatter to Q/K/V [b][h][s][hd] as bf16.
#pragma unroll
    for (int i = 0; i < 2; ++i)
#pragma unroll
        for (int j = 0; j < 4; ++j) {
            const int n = nBase + wc * 64 + j * 16 + lidx;   // 0..3071
            const float bv = bias[n];
            const int which = n >> 10;                        // 0:q 1:k 2:v
            const int rem = n & 1023;
            const int h  = rem >> 6;
            const int hd = rem & 63;
            unsigned short* dst = (which == 0) ? Qb : ((which == 1) ? Kb : Vb);
#pragma unroll
            for (int r = 0; r < 8; ++r) {
                const int m = mBase + wr * 32 + i * 16 + half * 8 + r;
                const int b = m >> 11;
                const int s = m & 2047;
                dst[((((size_t)b * 16 + h) * 2048 + s) << 6) + hd] =
                    f32_to_bf16(acc[i][j][r] + bv);
            }
        }
}

// ---------------------------------------------------------------------------
// Flash attention: one block per (b*h, q-tile of 128). 8 waves; each wave
// owns 16 q rows x 64 hd. Online softmax over 32 key-blocks of 64.
// ---------------------------------------------------------------------------
__global__ __launch_bounds__(256) void attn_kernel(
    const unsigned short* __restrict__ Qb,
    const unsigned short* __restrict__ Kb,
    const unsigned short* __restrict__ Vb,
    unsigned short* __restrict__ Ob)       // [b][s][h][hd] bf16 = [8192][1024]
{
    __shared__ unsigned short Qs[128][64];     // [qrow][hd]
    __shared__ unsigned short Ks[64][64];      // [key][hd]
    __shared__ unsigned short Vt[64][64];      // [hd][key] (transposed)
    __shared__ unsigned short Ps[8][16][64];   // per-wave P tile [qrow][key]

    const int tid  = threadIdx.x;
    const int wave = tid >> 5;
    const int lane = tid & 31;
    const int half = lane >> 4;
    const int lidx = lane & 15;
    const int bh    = blockIdx.x;              // b*16 + h
    const int qBase = blockIdx.y * 128;
    const size_t headOff = (size_t)bh * 2048 * 64;

    // Load Q tile (128x64 bf16 = 1024 uint4, 4 per thread).
    {
        const uint4* Qg4 = (const uint4*)(Qb + headOff + (size_t)qBase * 64);
#pragma unroll
        for (int i = 0; i < 4; ++i) {
            int s = i * 256 + tid;
            int r = s >> 3, c = s & 7;          // row, uint4 col
            ((uint4*)Qs)[s] = Qg4[r * 8 + c];
        }
    }
    __syncthreads();

    FragU aq[2];
    {
        const int row = wave * 16 + lidx;
#pragma unroll
        for (int t = 0; t < 2; ++t)
#pragma unroll
            for (int p = 0; p < 8; ++p)
                aq[t].u[p] = *(const unsigned*)&Qs[row][t * 32 + frag_kk(p, half)];
    }

    v8f o[4];
    for (int t = 0; t < 4; ++t)
        for (int e = 0; e < 8; ++e) o[t][e] = 0.0f;
    float mrow[8], lrow[8];
    for (int r = 0; r < 8; ++r) { mrow[r] = -1e30f; lrow[r] = 0.0f; }

    const float scale = 0.125f;  // 1/sqrt(64)

    for (int kb = 0; kb < 2048; kb += 64) {
        __syncthreads();
        // Stage K tile [key][hd] (512 uint4) and V tile transposed [hd][key].
        {
            const uint4* Kg4 = (const uint4*)(Kb + headOff + (size_t)kb * 64);
#pragma unroll
            for (int i = 0; i < 2; ++i) {
                int s = i * 256 + tid;
                int r = s >> 3, c = s & 7;
                ((uint4*)Ks)[s] = Kg4[r * 8 + c];
            }
            const unsigned* Vg = (const unsigned*)(Vb + headOff + (size_t)kb * 64);
#pragma unroll
            for (int i = 0; i < 8; ++i) {
                int f = i * 256 + tid;
                int r = f >> 5, c = f & 31;
                unsigned w = Vg[r * 32 + c];
                Vt[2 * c][r]     = (unsigned short)(w & 0xffffu);
                Vt[2 * c + 1][r] = (unsigned short)(w >> 16);
            }
        }
        // Prefetch next key-block of K/V into cache (global_prefetch_b8).
        if (kb + 64 < 2048) {
            __builtin_prefetch(Kb + headOff + (size_t)(kb + 64) * 64 + tid * 16, 0, 1);
            __builtin_prefetch(Vb + headOff + (size_t)(kb + 64) * 64 + tid * 16, 0, 1);
        }
        __syncthreads();

        // S = scale * Q @ K^T  (16q x 64k per wave, 4 C-fragments)
        v8f s[4];
        for (int j = 0; j < 4; ++j)
            for (int e = 0; e < 8; ++e) s[j][e] = 0.0f;
#pragma unroll
        for (int j = 0; j < 4; ++j) {
            const int key = j * 16 + lidx;
#pragma unroll
            for (int t = 0; t < 2; ++t) {
                FragU bk;
#pragma unroll
                for (int p = 0; p < 8; ++p)
                    bk.u[p] = *(const unsigned*)&Ks[key][t * 32 + frag_kk(p, half)];
                s[j] = __builtin_amdgcn_wmma_f32_16x16x32_bf16(
                    false, aq[t].v, false, bk.v, (short)0, s[j], false, false);
            }
        }

        // Online softmax update (rows in VGPR index r; cols across the 16
        // lanes of each half -> shfl_xor masks 1,2,4,8 reduce within half).
#pragma unroll
        for (int j = 0; j < 4; ++j)
            for (int e = 0; e < 8; ++e) s[j][e] *= scale;

#pragma unroll
        for (int r = 0; r < 8; ++r) {
            float mx = fmaxf(fmaxf(s[0][r], s[1][r]), fmaxf(s[2][r], s[3][r]));
            for (int d = 1; d < 16; d <<= 1)
                mx = fmaxf(mx, __shfl_xor(mx, d, 32));
            const float newm = fmaxf(mrow[r], mx);
            const float corr = __expf(mrow[r] - newm);
            mrow[r] = newm;
            float sum = 0.0f;
#pragma unroll
            for (int j = 0; j < 4; ++j) {
                s[j][r] = __expf(s[j][r] - newm);
                sum += s[j][r];
            }
            for (int d = 1; d < 16; d <<= 1)
                sum += __shfl_xor(sum, d, 32);
            lrow[r] = lrow[r] * corr + sum;
#pragma unroll
            for (int t = 0; t < 4; ++t) o[t][r] *= corr;
        }

        // Stage P (C-layout -> LDS) so we can regather in A-fragment layout.
#pragma unroll
        for (int j = 0; j < 4; ++j)
#pragma unroll
            for (int r = 0; r < 8; ++r)
                Ps[wave][half * 8 + r][j * 16 + lidx] = f32_to_bf16(s[j][r]);
        __syncthreads();

        // O += P @ V   (contraction over 64 keys = 2 WMMA steps per hd tile)
        FragU ap[2];
#pragma unroll
        for (int u = 0; u < 2; ++u)
#pragma unroll
            for (int p = 0; p < 8; ++p)
                ap[u].u[p] = *(const unsigned*)&Ps[wave][lidx][u * 32 + frag_kk(p, half)];

#pragma unroll
        for (int t = 0; t < 4; ++t) {
            const int hd = t * 16 + lidx;
#pragma unroll
            for (int u = 0; u < 2; ++u) {
                FragU bv;
#pragma unroll
                for (int p = 0; p < 8; ++p)
                    bv.u[p] = *(const unsigned*)&Vt[hd][u * 32 + frag_kk(p, half)];
                o[t] = __builtin_amdgcn_wmma_f32_16x16x32_bf16(
                    false, ap[u].v, false, bv.v, (short)0, o[t], false, false);
            }
        }
    }

    // Normalize and store to [b][s][h][hd] bf16.
    const int b = bh >> 4, h = bh & 15;
#pragma unroll
    for (int t = 0; t < 4; ++t) {
        const int hd = t * 16 + lidx;
#pragma unroll
        for (int r = 0; r < 8; ++r) {
            const int qrow = qBase + wave * 16 + half * 8 + r;
            const float v = o[t][r] / lrow[r];
            Ob[(((size_t)b * 2048 + qrow) * 16 + h) * 64 + hd] = f32_to_bf16(v);
        }
    }
}

// ---------------------------------------------------------------------------
// GEMM 2: output projection. out[8192,1024] = Attn[8192,1024] @ Wout^T + b.
// Same 128x128 tiling; fp32 epilogue straight to d_out.
// ---------------------------------------------------------------------------
__global__ __launch_bounds__(256) void gemm_out_kernel(
    const unsigned short* __restrict__ A,     // [8192][1024] bf16 (attn out)
    const unsigned short* __restrict__ W,     // [1024][1024] bf16
    const float* __restrict__ bias,           // [1024]
    float* __restrict__ Out)                  // [8192][1024] fp32
{
    __shared__ unsigned short As[128][32];
    __shared__ unsigned short Bs[128][32];

    const int tid  = threadIdx.x;
    const int wave = tid >> 5;
    const int lane = tid & 31;
    const int half = lane >> 4;
    const int lidx = lane & 15;
    const int wr   = wave & 3;
    const int wc   = wave >> 2;
    const int mBase = blockIdx.x * 128;
    const int nBase = blockIdx.y * 128;

    v8f acc[2][4];
    for (int i = 0; i < 2; ++i)
        for (int j = 0; j < 4; ++j)
            for (int e = 0; e < 8; ++e) acc[i][j][e] = 0.0f;

    const uint4* Ag4 = (const uint4*)A;
    const uint4* Wg4 = (const uint4*)W;

    for (int kb = 0; kb < 1024; kb += 32) {
        __syncthreads();
#pragma unroll
        for (int i = 0; i < 2; ++i) {
            int s = i * 256 + tid;
            int r = s >> 2, c = s & 3;
            ((uint4*)As)[s] = Ag4[(size_t)(mBase + r) * 128 + (kb >> 3) + c];
            ((uint4*)Bs)[s] = Wg4[(size_t)(nBase + r) * 128 + (kb >> 3) + c];
        }
        __syncthreads();

        FragU af[2];
#pragma unroll
        for (int i = 0; i < 2; ++i) {
            const int arow = wr * 32 + i * 16 + lidx;
#pragma unroll
            for (int p = 0; p < 8; ++p)
                af[i].u[p] = *(const unsigned*)&As[arow][frag_kk(p, half)];
        }
#pragma unroll
        for (int j = 0; j < 4; ++j) {
            FragU bf;
            const int brow = wc * 64 + j * 16 + lidx;
#pragma unroll
            for (int p = 0; p < 8; ++p)
                bf.u[p] = *(const unsigned*)&Bs[brow][frag_kk(p, half)];
#pragma unroll
            for (int i = 0; i < 2; ++i)
                acc[i][j] = __builtin_amdgcn_wmma_f32_16x16x32_bf16(
                    false, af[i].v, false, bf.v, (short)0, acc[i][j], false, false);
        }
    }

#pragma unroll
    for (int i = 0; i < 2; ++i)
#pragma unroll
        for (int j = 0; j < 4; ++j) {
            const int n = nBase + wc * 64 + j * 16 + lidx;
            const float bv = bias[n];
#pragma unroll
            for (int r = 0; r < 8; ++r) {
                const int m = mBase + wr * 32 + i * 16 + half * 8 + r;
                Out[(size_t)m * 1024 + n] = acc[i][j][r] + bv;
            }
        }
}

// ---------------------------------------------------------------------------
// Host-side launcher
// ---------------------------------------------------------------------------
extern "C" void kernel_launch(void* const* d_in, const int* in_sizes, int n_in,
                              void* d_out, int out_size, void* d_ws, size_t ws_size,
                              hipStream_t stream) {
    const float* x      = (const float*)d_in[0];  // [4,2048,1024]
    const float* qkv_w  = (const float*)d_in[1];  // [3072,1024]
    const float* qkv_b  = (const float*)d_in[2];  // [3072]
    const float* out_w  = (const float*)d_in[3];  // [1024,1024]
    const float* out_b  = (const float*)d_in[4];  // [1024]
    float* out = (float*)d_out;

    const size_t M   = 8192;       // 4*2048 tokens
    const size_t D   = 1024;
    const size_t N3  = 3072;

    unsigned short* ws   = (unsigned short*)d_ws;
    unsigned short* Xbf  = ws;                    // M*D
    unsigned short* Wqkv = Xbf  + M * D;          // N3*D
    unsigned short* Wout = Wqkv + N3 * D;         // D*D
    unsigned short* Qb   = Wout + D * D;          // M*D   ([b][h][s][hd])
    unsigned short* Kb   = Qb   + M * D;          // M*D
    unsigned short* Vb   = Kb   + M * D;          // M*D
    unsigned short* Attn = Vb   + M * D;          // M*D   ([b][s][h][hd])

    // 1. fp32 -> bf16 conversions
    {
        int n;
        n = (int)(M * D);
        cvt_f32_bf16_kernel<<<(n + 255) / 256, 256, 0, stream>>>(x, Xbf, n);
        n = (int)(N3 * D);
        cvt_f32_bf16_kernel<<<(n + 255) / 256, 256, 0, stream>>>(qkv_w, Wqkv, n);
        n = (int)(D * D);
        cvt_f32_bf16_kernel<<<(n + 255) / 256, 256, 0, stream>>>(out_w, Wout, n);
    }

    // 2. QKV projection GEMM (WMMA), scatter to per-head Q/K/V
    gemm_qkv_kernel<<<dim3(64, 24), 256, 0, stream>>>(Xbf, Wqkv, qkv_b,
                                                      Qb, Kb, Vb);

    // 3. Flash attention (WMMA + online softmax), 128-row q-tiles
    attn_kernel<<<dim3(64, 16), 256, 0, stream>>>(Qb, Kb, Vb, Attn);

    // 4. Output projection GEMM (WMMA), fp32 epilogue
    gemm_out_kernel<<<dim3(64, 8), 256, 0, stream>>>(Attn, Wout, out_b, out);
}
